// KeywordMambaNet_78176994722581
// MI455X (gfx1250) — compile-verified, matrix operand order checked by reading, not run on hardware
//
#include <hip/hip_runtime.h>
#include <hip/hip_bf16.h>

#define DMD   128
#define NLAY  4
#define NSS   16
#define KCONV 4
#define EDD   256
#define DTRR  8
#define BB    16
#define SEQ   500
#define NMCH  80
#define NCMD  35
#define EPSV  1e-5f
#define MROWS (BB*SEQ)     // 8000
#define NPAD  48

typedef _Float16 v16h __attribute__((ext_vector_type(16)));
typedef _Float16 v8h  __attribute__((ext_vector_type(8)));
typedef float    v8f  __attribute__((ext_vector_type(8)));

__device__ __forceinline__ float siluf(float v) { return v / (1.f + __expf(-v)); }

// Load one 16x32 f16 fragment (A-style or B-style, both are K-contiguous rows).
// lane<16: K-chunks {k .. k+7, k+16 .. k+23}; lane>=16: {k+8..k+15, k+24..k+31}
// (the +8 for hi lanes is folded into the row pointer by the caller).
__device__ __forceinline__ v16h load_frag(const _Float16* __restrict__ p) {
    v8h lo = *(const v8h*)p;
    v8h hi = *(const v8h*)(p + 16);
    v16h r;
#pragma unroll
    for (int i = 0; i < 8; ++i) { r[i] = lo[i]; r[i + 8] = hi[i]; }
    return r;
}

// ---------------------------------------------------------------------------
// Stem: conv1d(NM=80 -> DM=128, k=5, pad 2) + BN + silu, output (B,L,DM) f32
// ---------------------------------------------------------------------------
__global__ void stem_kernel(const float* __restrict__ feat, const float* __restrict__ w,
                            const float* __restrict__ bias, const float* __restrict__ gamma,
                            const float* __restrict__ beta, const float* __restrict__ mean,
                            const float* __restrict__ var, float* __restrict__ x) {
    int idx = blockIdx.x * blockDim.x + threadIdx.x;
    if (idx >= BB * SEQ * DMD) return;
    int d = idx % DMD;
    int l = (idx / DMD) % SEQ;
    int b = idx / (DMD * SEQ);
    float s = bias[d];
    for (int nm = 0; nm < NMCH; ++nm) {
        const float* frow = feat + ((size_t)b * NMCH + nm) * SEQ;
        const float* wrow = w + ((size_t)d * NMCH + nm) * 5;
#pragma unroll
        for (int j = 0; j < 5; ++j) {
            int ls = l - 2 + j;
            if (ls >= 0 && ls < SEQ) s += wrow[j] * frow[ls];
        }
    }
    s = (s - mean[d]) * rsqrtf(var[d] + EPSV);
    s = s * gamma[d] + beta[d];
    x[((size_t)b * SEQ + l) * DMD + d] = siluf(s);
}

// ---------------------------------------------------------------------------
// Per-layer weight conversion to f16 (x_proj padded 40->48 rows with zeros)
// ---------------------------------------------------------------------------
__global__ void convert_weights_kernel(const float* __restrict__ inW, const float* __restrict__ xpW,
                                       const float* __restrict__ outW, _Float16* __restrict__ in16,
                                       _Float16* __restrict__ xp16, _Float16* __restrict__ out16) {
    int idx = blockIdx.x * blockDim.x + threadIdx.x;
    const int N_IN = 2 * EDD * DMD;   // 512*128
    const int N_XP = NPAD * EDD;      // 48*256
    const int N_OUT = DMD * EDD;      // 128*256
    if (idx < N_IN) { in16[idx] = (_Float16)inW[idx]; return; }
    idx -= N_IN;
    if (idx < N_XP) {
        int n = idx / EDD, k = idx % EDD;
        xp16[idx] = (n < (DTRR + 2 * NSS)) ? (_Float16)xpW[n * EDD + k] : (_Float16)0.f;
        return;
    }
    idx -= N_XP;
    if (idx < N_OUT) out16[idx] = (_Float16)outW[idx];
}

// ---------------------------------------------------------------------------
// RMSNorm over DM=128 per row, write f16 activations for WMMA
// ---------------------------------------------------------------------------
__global__ __launch_bounds__(128) void rmsnorm_kernel(const float* __restrict__ x,
                                                      const float* __restrict__ w,
                                                      _Float16* __restrict__ u16) {
    int m = blockIdx.x, t = threadIdx.x;
    __shared__ float red[DMD];
    float v = x[(size_t)m * DMD + t];
    red[t] = v * v;
    __syncthreads();
    for (int s = 64; s > 0; s >>= 1) {
        if (t < s) red[t] += red[t + s];
        __syncthreads();
    }
    float sc = rsqrtf(red[0] / DMD + EPSV);
    u16[(size_t)m * DMD + t] = (_Float16)(v * sc * w[t]);
}

// ---------------------------------------------------------------------------
// Register-blocked WMMA GEMM: one wave computes a 32 x (16*NF) macro-tile of
// C[m][n] (+= if residual) = sum_k A[m][k] * B[n][k]
// A: MxK f16 row-major, B: NxK f16 row-major (both K-contiguous -> direct
// fragment loads per the CDNA5 16-bit layout). Per K-step: 2 A-frags + NF
// B-frags feed 2*NF v_wmma ops (A reused NF x, B reused 2x in registers).
// M multiple of 32, N multiple of 16*NF, K multiple of 32. EXEC all-ones.
// ---------------------------------------------------------------------------
template <int NF>
__global__ __launch_bounds__(32) void wmma_gemm_kernel(const _Float16* __restrict__ A,
                                                       const _Float16* __restrict__ Bm,
                                                       float* __restrict__ C,
                                                       int K, int ldc, int residual) {
    int lane = threadIdx.x;                 // 0..31, wave32
    int m0 = blockIdx.y * 32;
    int n0 = blockIdx.x * (16 * NF);
    int koff = (lane >> 4) << 3;            // hi half-lanes start at K+8
    const _Float16* arow0 = A + (size_t)(m0 + (lane & 15)) * K + koff;
    const _Float16* arow1 = arow0 + (size_t)16 * K;
    const _Float16* brow[NF];
#pragma unroll
    for (int nf = 0; nf < NF; ++nf)
        brow[nf] = Bm + (size_t)(n0 + nf * 16 + (lane & 15)) * K + koff;

    v8f acc[2][NF];
#pragma unroll
    for (int nf = 0; nf < NF; ++nf) { acc[0][nf] = (v8f){}; acc[1][nf] = (v8f){}; }

    for (int k0 = 0; k0 < K; k0 += 32) {
        if (k0 + 32 < K) {                  // global_prefetch_b8 next K chunk
            __builtin_prefetch(arow0 + k0 + 32, 0, 3);
            __builtin_prefetch(arow1 + k0 + 32, 0, 3);
        }
        v16h a0 = load_frag(arow0 + k0);
        v16h a1 = load_frag(arow1 + k0);
#pragma unroll
        for (int nf = 0; nf < NF; ++nf) {
            v16h b = load_frag(brow[nf] + k0);
            acc[0][nf] = __builtin_amdgcn_wmma_f32_16x16x32_f16(false, a0, false, b,
                                                               (short)0, acc[0][nf], false, false);
            acc[1][nf] = __builtin_amdgcn_wmma_f32_16x16x32_f16(false, a1, false, b,
                                                               (short)0, acc[1][nf], false, false);
        }
    }
    // C/D layout: VGPR v -> M = v (lanes 0-15) or v+8 (lanes 16-31); N = lane&15
#pragma unroll
    for (int mf = 0; mf < 2; ++mf) {
        int rbase = m0 + mf * 16 + ((lane >> 4) << 3);
#pragma unroll
        for (int nf = 0; nf < NF; ++nf) {
            int col = n0 + nf * 16 + (lane & 15);
#pragma unroll
            for (int v = 0; v < 8; ++v) {
                float* p = C + (size_t)(rbase + v) * ldc + col;
                float val = acc[mf][nf][v];
                if (residual) val += *p;
                *p = val;
            }
        }
    }
}

// ---------------------------------------------------------------------------
// Depthwise causal conv (K=4) over L + bias + silu; input = xz[:, :ED]
// ---------------------------------------------------------------------------
__global__ void dwconv_kernel(const float* __restrict__ xz, const float* __restrict__ cw,
                              const float* __restrict__ cb, float* __restrict__ xin,
                              _Float16* __restrict__ xin16) {
    int idx = blockIdx.x * blockDim.x + threadIdx.x;
    if (idx >= BB * SEQ * EDD) return;
    int e = idx % EDD;
    int l = (idx / EDD) % SEQ;
    int b = idx / (EDD * SEQ);
    float s = cb[e];
    const float* w = cw + e * KCONV;
#pragma unroll
    for (int j = 0; j < KCONV; ++j) {
        int ls = l - (KCONV - 1) + j;
        if (ls >= 0) s += w[j] * xz[((size_t)b * SEQ + ls) * (2 * EDD) + e];
    }
    s = siluf(s);
    size_t m = (size_t)b * SEQ + l;
    xin[m * EDD + e] = s;
    xin16[m * EDD + e] = (_Float16)s;
}

// ---------------------------------------------------------------------------
// dt_proj (K=8) + softplus -> delta
// ---------------------------------------------------------------------------
__global__ void dt_kernel(const float* __restrict__ dbc, const float* __restrict__ dtW,
                          const float* __restrict__ dtb, float* __restrict__ delta) {
    int idx = blockIdx.x * blockDim.x + threadIdx.x;
    if (idx >= MROWS * EDD) return;
    int e = idx % EDD;
    size_t m = (size_t)(idx / EDD);
    float s = dtb[e];
    const float* r = dbc + m * NPAD;
    const float* w = dtW + (size_t)e * DTRR;
#pragma unroll
    for (int j = 0; j < DTRR; ++j) s += r[j] * w[j];
    delta[idx] = (s > 20.f) ? s : log1pf(__expf(s));
}

// ---------------------------------------------------------------------------
// Selective scan: one block per batch b (256 threads = one channel e each),
// NS=16 state in registers, B/C vectors broadcast via LDS each timestep.
// Fuses y = h.C + D*xin, gate by silu(z), and writes f16 for out_proj WMMA.
// ---------------------------------------------------------------------------
__global__ __launch_bounds__(256) void scan_kernel(const float* __restrict__ delta,
                                                   const float* __restrict__ xin,
                                                   const float* __restrict__ dbc,
                                                   const float* __restrict__ xz,
                                                   const float* __restrict__ A_log,
                                                   const float* __restrict__ Dp,
                                                   _Float16* __restrict__ y16) {
    int b = blockIdx.x;
    int e = threadIdx.x;              // 0..255
    __shared__ float sB[NSS], sC[NSS];
    float Arow[NSS];
#pragma unroll
    for (int n = 0; n < NSS; ++n) Arow[n] = -__expf(A_log[(size_t)e * NSS + n]);
    float Dv = Dp[e];
    float h[NSS];
#pragma unroll
    for (int n = 0; n < NSS; ++n) h[n] = 0.f;
    for (int l = 0; l < SEQ; ++l) {
        size_t m = (size_t)b * SEQ + l;
        if (e < 2 * NSS) {
            float v = dbc[m * NPAD + DTRR + e];
            if (e < NSS) sB[e] = v; else sC[e - NSS] = v;
        }
        __syncthreads();
        float d = delta[m * EDD + e];
        float xv = xin[m * EDD + e];
        float dx = d * xv;
        float acc = Dv * xv;
#pragma unroll
        for (int n = 0; n < NSS; ++n) {
            float dA = __expf(d * Arow[n]);
            h[n] = dA * h[n] + dx * sB[n];
            acc += h[n] * sC[n];
        }
        float zv = xz[m * (2 * EDD) + EDD + e];
        y16[m * EDD + e] = (_Float16)(acc * siluf(zv));
        __syncthreads();
    }
}

// ---------------------------------------------------------------------------
// Final LayerNorm over DM per row (in place)
// ---------------------------------------------------------------------------
__global__ __launch_bounds__(128) void ln_kernel(float* __restrict__ x, const float* __restrict__ w,
                                                 const float* __restrict__ bvec) {
    int m = blockIdx.x, t = threadIdx.x;
    __shared__ float s1[DMD], s2[DMD];
    float v = x[(size_t)m * DMD + t];
    s1[t] = v; s2[t] = v * v;
    __syncthreads();
    for (int s = 64; s > 0; s >>= 1) {
        if (t < s) { s1[t] += s1[t + s]; s2[t] += s2[t + s]; }
        __syncthreads();
    }
    float mu = s1[0] / DMD;
    float var = s2[0] / DMD - mu * mu;
    x[(size_t)m * DMD + t] = (v - mu) * rsqrtf(var + EPSV) * w[t] + bvec[t];
}

// ---------------------------------------------------------------------------
// Mean pool over L
// ---------------------------------------------------------------------------
__global__ void pool_kernel(const float* __restrict__ x, float* __restrict__ pooled) {
    int idx = blockIdx.x * blockDim.x + threadIdx.x;
    if (idx >= BB * DMD) return;
    int b = idx / DMD, d = idx % DMD;
    float s = 0.f;
    for (int l = 0; l < SEQ; ++l) s += x[((size_t)b * SEQ + l) * DMD + d];
    pooled[idx] = s / (float)SEQ;
}

// ---------------------------------------------------------------------------
// Head: emb = silu(pooled @ embW^T + b); cmd/wake logits. One block per batch.
// ---------------------------------------------------------------------------
__global__ __launch_bounds__(128) void head_kernel(const float* __restrict__ pooled,
                                                   const float* __restrict__ embW, const float* __restrict__ embB,
                                                   const float* __restrict__ cmdW, const float* __restrict__ cmdB,
                                                   const float* __restrict__ wakeW, const float* __restrict__ wakeB,
                                                   float* __restrict__ out_cmd, float* __restrict__ out_wake,
                                                   float* __restrict__ out_emb) {
    int b = blockIdx.x, t = threadIdx.x;
    __shared__ float sp[DMD], se[DMD];
    sp[t] = pooled[b * DMD + t];
    __syncthreads();
    float s = embB[t];
    for (int k = 0; k < DMD; ++k) s += sp[k] * embW[t * DMD + k];
    float e = siluf(s);
    se[t] = e;
    out_emb[b * DMD + t] = e;
    __syncthreads();
    if (t < NCMD) {
        float c = cmdB[t];
        for (int k = 0; k < DMD; ++k) c += se[k] * cmdW[t * DMD + k];
        out_cmd[b * NCMD + t] = c;
    }
    if (t == 0) {
        float wv = wakeB[0];
        for (int k = 0; k < DMD; ++k) wv += se[k] * wakeW[k];
        out_wake[b] = wv;
    }
}

// ---------------------------------------------------------------------------
extern "C" void kernel_launch(void* const* d_in, const int* in_sizes, int n_in,
                              void* d_out, int out_size, void* d_ws, size_t ws_size,
                              hipStream_t stream) {
    const float* features = (const float*)d_in[0];
    const float* stem_w   = (const float*)d_in[1];
    const float* stem_b   = (const float*)d_in[2];
    const float* bn_gamma = (const float*)d_in[3];
    const float* bn_beta  = (const float*)d_in[4];
    const float* bn_mean  = (const float*)d_in[5];
    const float* bn_var   = (const float*)d_in[6];
    const float* norm_w   = (const float*)d_in[7];
    const float* in_proj  = (const float*)d_in[8];
    const float* conv_w   = (const float*)d_in[9];
    const float* conv_b   = (const float*)d_in[10];
    const float* x_proj   = (const float*)d_in[11];
    const float* dt_W     = (const float*)d_in[12];
    const float* dt_b     = (const float*)d_in[13];
    const float* A_log    = (const float*)d_in[14];
    const float* Dparam   = (const float*)d_in[15];
    const float* out_proj = (const float*)d_in[16];
    const float* ln_w     = (const float*)d_in[17];
    const float* ln_b     = (const float*)d_in[18];
    const float* emb_W    = (const float*)d_in[19];
    const float* emb_b    = (const float*)d_in[20];
    const float* cmd_W    = (const float*)d_in[21];
    const float* cmd_b    = (const float*)d_in[22];
    const float* wake_W   = (const float*)d_in[23];
    const float* wake_b   = (const float*)d_in[24];

    char* ws = (char*)d_ws;
    size_t off = 0;
    auto alloc = [&](size_t bytes) -> void* {
        void* p = ws + off;
        off += (bytes + 255) & ~(size_t)255;
        return p;
    };
    float*     x      = (float*)alloc((size_t)MROWS * DMD * 4);
    _Float16*  u16    = (_Float16*)alloc((size_t)MROWS * DMD * 2);
    float*     xz     = (float*)alloc((size_t)MROWS * 2 * EDD * 4);
    float*     xin    = (float*)alloc((size_t)MROWS * EDD * 4);
    _Float16*  xin16  = (_Float16*)alloc((size_t)MROWS * EDD * 2);
    float*     dbc    = (float*)alloc((size_t)MROWS * NPAD * 4);
    float*     delta  = (float*)alloc((size_t)MROWS * EDD * 4);
    _Float16*  y16    = (_Float16*)alloc((size_t)MROWS * EDD * 2);
    float*     pooled = (float*)alloc((size_t)BB * DMD * 4);
    _Float16*  w_in16 = (_Float16*)alloc((size_t)2 * EDD * DMD * 2);
    _Float16*  w_xp16 = (_Float16*)alloc((size_t)NPAD * EDD * 2);
    _Float16*  w_o16  = (_Float16*)alloc((size_t)DMD * EDD * 2);

    float* out_cmd  = (float*)d_out;                  // 16*35
    float* out_wake = (float*)d_out + BB * NCMD;      // 16
    float* out_emb  = (float*)d_out + BB * NCMD + BB; // 16*128

    // Stem
    {
        int n = BB * SEQ * DMD;
        stem_kernel<<<(n + 255) / 256, 256, 0, stream>>>(features, stem_w, stem_b,
                                                         bn_gamma, bn_beta, bn_mean, bn_var, x);
    }

    const int NW = 2 * EDD * DMD + NPAD * EDD + DMD * EDD;   // convert thread count
    for (int i = 0; i < NLAY; ++i) {
        convert_weights_kernel<<<(NW + 255) / 256, 256, 0, stream>>>(
            in_proj + (size_t)i * 2 * EDD * DMD,
            x_proj + (size_t)i * (DTRR + 2 * NSS) * EDD,
            out_proj + (size_t)i * DMD * EDD,
            w_in16, w_xp16, w_o16);

        rmsnorm_kernel<<<MROWS, 128, 0, stream>>>(x, norm_w + (size_t)i * DMD, u16);

        // in_proj: M=8000, N=512, K=128 -> 32x64 macro-tiles
        wmma_gemm_kernel<4><<<dim3(2 * EDD / 64, MROWS / 32), 32, 0, stream>>>(
            u16, w_in16, xz, DMD, 2 * EDD, 0);

        {
            int n = BB * SEQ * EDD;
            dwconv_kernel<<<(n + 255) / 256, 256, 0, stream>>>(
                xz, conv_w + (size_t)i * EDD * KCONV, conv_b + (size_t)i * EDD, xin, xin16);
        }

        // x_proj: M=8000, N=48(pad), K=256 -> 32x48 macro-tiles
        wmma_gemm_kernel<3><<<dim3(1, MROWS / 32), 32, 0, stream>>>(
            xin16, w_xp16, dbc, EDD, NPAD, 0);

        {
            int n = MROWS * EDD;
            dt_kernel<<<(n + 255) / 256, 256, 0, stream>>>(
                dbc, dt_W + (size_t)i * EDD * DTRR, dt_b + (size_t)i * EDD, delta);
        }

        scan_kernel<<<BB, EDD, 0, stream>>>(delta, xin, dbc, xz,
                                            A_log + (size_t)i * EDD * NSS,
                                            Dparam + (size_t)i * EDD, y16);

        // out_proj + residual: M=8000, N=128, K=256 -> 32x64 macro-tiles
        wmma_gemm_kernel<4><<<dim3(DMD / 64, MROWS / 32), 32, 0, stream>>>(
            y16, w_o16, x, EDD, DMD, 1);
    }

    ln_kernel<<<MROWS, 128, 0, stream>>>(x, ln_w, ln_b);
    {
        int n = BB * DMD;
        pool_kernel<<<(n + 255) / 256, 256, 0, stream>>>(x, pooled);
    }
    head_kernel<<<BB, 128, 0, stream>>>(pooled, emb_W, emb_b, cmd_W, cmd_b,
                                        wake_W, wake_b, out_cmd, out_wake, out_emb);
}
